// PostBackboneRefineAndGate_14645838479427
// MI455X (gfx1250) — matrix operand order repaired and checked by reading, not compiled, not run on hardware
//
#include <hip/hip_runtime.h>
#include <hip/hip_bf16.h>
#include <math.h>
#include <stdint.h>

// MI455X / gfx1250, wave32. bf16 WMMA 16x16x32; async global->LDS + TDM staging.
// B=32, L=1024, H=1024, HC=256.

#define B_  32
#define L_  1024
#define H_  1024
#define HC_ 256

typedef __attribute__((ext_vector_type(16))) __bf16 bf16x16;
typedef __attribute__((ext_vector_type(8)))  __bf16 bf16x8;
typedef __attribute__((ext_vector_type(8)))  float  f32x8;
typedef __attribute__((ext_vector_type(4)))  float  f32x4;
typedef __attribute__((ext_vector_type(4)))  unsigned int u32x4;
typedef __attribute__((ext_vector_type(8)))  int    i32x8;
typedef __attribute__((ext_vector_type(4)))  int    i32x4;

__device__ __forceinline__ bf16x16 cat8(bf16x8 lo, bf16x8 hi) {
  return __builtin_shufflevector(lo, hi, 0,1,2,3,4,5,6,7,8,9,10,11,12,13,14,15);
}

// A-tile (16x32 bf16) from f32 row-major source; per ISA layout:
// lane<16: M=lane, K = k0+{0..7, 16..23}; lane>=16: M=lane-16, K = k0+{8..15, 24..31}
__device__ __forceinline__ bf16x16 loadA_f32(const float* __restrict__ base, int lda,
                                             int row, int k0, int lane) {
  const int kh = (lane >> 4) << 3;
  const float* p = base + (long)row * lda + k0 + kh;
  f32x4 a0 = *(const f32x4*)(p);
  f32x4 a1 = *(const f32x4*)(p + 4);
  f32x4 b0 = *(const f32x4*)(p + 16);
  f32x4 b1 = *(const f32x4*)(p + 20);
  bf16x16 r;
#pragma unroll
  for (int i = 0; i < 4; ++i) {
    r[i]      = (__bf16)a0[i];
    r[i + 4]  = (__bf16)a1[i];
    r[i + 8]  = (__bf16)b0[i];
    r[i + 12] = (__bf16)b1[i];
  }
  return r;
}

__device__ __forceinline__ bf16x16 loadA_bf16(const __bf16* __restrict__ base, int lda,
                                              int row, int k0, int lane) {
  const int kh = (lane >> 4) << 3;
  const __bf16* p = base + (long)row * lda + k0 + kh;
  bf16x8 lo = *(const bf16x8*)(p);
  bf16x8 hi = *(const bf16x8*)(p + 16);
  return cat8(lo, hi);
}

// B-tile (32x16 bf16) from weights stored [N, K] row-major:
// lane gives N (lane&15); lane<16: K = k0+{0..7, 16..23}; lane>=16: K = k0+{8..15, 24..31}.
__device__ __forceinline__ bf16x16 loadB(const __bf16* __restrict__ Wt, int ldk,
                                         int n0, int k0, int lane) {
  const int n  = n0 + (lane & 15);
  const int kg = (lane >> 4) << 3;
  const __bf16* p = Wt + (long)n * ldk + k0 + kg;
  bf16x8 lo = *(const bf16x8*)(p);
  bf16x8 hi = *(const bf16x8*)(p + 16);
  return cat8(lo, hi);
}

// Branch-free gelu (tanh approx): tanh(y) = 1 - 2/(exp(2y)+1); exact saturation at +-inf.
__device__ __forceinline__ float gelu_f(float x) {
  float y = 0.7978845608028654f * (x + 0.044715f * x * x * x);
  float e = __expf(2.0f * y);
  float t = 1.0f - 2.0f / (e + 1.0f);
  return 0.5f * x * (1.0f + t);
}

__device__ __forceinline__ float sigmoid_f(float x) {
  return 1.0f / (1.0f + __expf(-x));
}

__device__ __forceinline__ f32x8 zero8() {
  f32x8 z = {0.f,0.f,0.f,0.f,0.f,0.f,0.f,0.f};
  return z;
}

// ---- CDNA5 async global->LDS copy (VFLAT GV form) + asynccnt wait ----
__device__ __forceinline__ void async_copy_b128(void* lds_ptr, const void* gptr) {
  // LDS aperture: addr[31:0] of the flat pointer is the LDS byte address.
  unsigned loff = (unsigned)(uintptr_t)lds_ptr;
  asm volatile("global_load_async_to_lds_b128 %0, %1, off"
               :: "v"(loff), "v"(gptr)
               : "memory");
}
__device__ __forceinline__ void wait_async0() {
  asm volatile("s_wait_asynccnt 0" ::: "memory");
}

// ---- CDNA5 Tensor Data Mover: 2D tile (bf16) global -> LDS ----
// D# packing per CDNA5 ISA (2D, no padding/iterate/gather).
// This toolchain's builtin takes 6 args: (g0, g1, g2, g3, g_extra, cpol).
__device__ __forceinline__ void tdm_load_tile_2d_bf16(void* lds_ptr, const void* gptr,
                                                      unsigned tensor_d0, unsigned tensor_d1,
                                                      unsigned tile_d0, unsigned tile_d1,
                                                      unsigned row_stride_elems) {
  unsigned lds = (unsigned)(uintptr_t)lds_ptr;
  unsigned long long ga = (unsigned long long)(uintptr_t)gptr;
  u32x4 g0;
  g0[0] = 1u;                                                // count=1, user D#
  g0[1] = lds;                                               // lds_addr (bytes)
  g0[2] = (unsigned)(ga & 0xffffffffu);                      // global_addr[31:0]
  g0[3] = (unsigned)((ga >> 32) & 0x01ffffffu) | (2u << 30); // global_addr[56:32] | type=2
  i32x8 g1;
  g1[0] = (int)(1u << 16);                                   // data_size=1 -> 2 bytes
  g1[1] = (int)((tensor_d0 & 0xffffu) << 16);                // tensor_dim0[15:0] @ bits 63:48
  g1[2] = (int)(((tensor_d0 >> 16) & 0xffffu) |              // tensor_dim0[31:16]
                ((tensor_d1 & 0xffffu) << 16));              // tensor_dim1[15:0]
  g1[3] = (int)(((tensor_d1 >> 16) & 0xffffu) |              // tensor_dim1[31:16]
                ((tile_d0 & 0xffffu) << 16));                // tile_dim0
  g1[4] = (int)(tile_d1 & 0xffffu);                          // tile_dim1 (tile_dim2=0)
  g1[5] = (int)row_stride_elems;                             // tensor_dim0_stride[31:0]
  g1[6] = 0;                                                 // stride hi / dim1_stride lo
  g1[7] = 0;
  i32x4 gz4 = {0, 0, 0, 0};
  i32x8 gz8 = {0, 0, 0, 0, 0, 0, 0, 0};
  __builtin_amdgcn_tensor_load_to_lds(g0, g1, gz4, gz4, gz8, 0);
}

// -------- weight convert + transpose: out[N,K] bf16 <- in[K,N] f32 --------
__global__ __launch_bounds__(256) void f32_to_bf16_t(const float* __restrict__ in,
                                                     __bf16* __restrict__ out,
                                                     int K, int N) {
  long i = (long)blockIdx.x * blockDim.x + threadIdx.x;
  if (i >= (long)K * N) return;
  int n = (int)(i % N);
  int k = (int)(i / N);
  out[(long)n * K + k] = (__bf16)in[i];
}

#define KCH 128

// -------- GEMM1: C1 = gelu(u@Wc1[:H] + z@Wc1[H:] + bc1) -> bf16 [32768,256] --------
// 8 waves share one 64-col N-group; B staged via async global->LDS, double-buffered.
__global__ __launch_bounds__(256) void gemm_corr1(const float* __restrict__ u,
                                                  const float* __restrict__ z,
                                                  const __bf16* __restrict__ Wc1t, // [256,2048]
                                                  const float* __restrict__ bc1,
                                                  __bf16* __restrict__ C1) {
  __shared__ __bf16 Bs[2][64 * KCH];        // 2 * 16KB
  const int tid  = threadIdx.x;
  const int lane = tid & 31;
  const int wave = tid >> 5;
  const int mb = blockIdx.x >> 2;           // 0..255
  const int ng = blockIdx.x & 3;            // 0..3
  const int n0base = ng << 6;
  const int m0 = ((mb << 3) + wave) << 4;
  const int row = m0 + (lane & 15);
  const int lh = lane >> 4;
  const int NCHUNK = 16;                    // 2048 / 128

  f32x8 acc[4];
#pragma unroll
  for (int j = 0; j < 4; ++j) acc[j] = zero8();

  auto stage = [&](int c, int buf) {
    const int kc = c << 7;
#pragma unroll
    for (int i = 0; i < 4; ++i) {
      const int s  = (i << 8) + tid;        // 0..1023
      const int rn = s >> 4;                // 0..63
      const int sk = (s & 15) << 3;         // element offset 0,8,...,120
      async_copy_b128(&Bs[buf][rn * KCH + sk],
                      Wc1t + (long)(n0base + rn) * (2 * H_) + kc + sk);
    }
  };

  stage(0, 0);
  for (int c = 0; c < NCHUNK; ++c) {
    wait_async0();
    __syncthreads();
    if (c + 1 < NCHUNK) stage(c + 1, (c + 1) & 1);
    const int kc = c << 7;
    const float* Abase = (kc < H_) ? u : z;       // chunk never straddles the u/z split
    const int kb = (kc < H_) ? kc : kc - H_;
    const __bf16* Bbuf = &Bs[c & 1][0];
#pragma unroll
    for (int kt = 0; kt < 4; ++kt) {
      bf16x16 a = loadA_f32(Abase, H_, row, kb + (kt << 5), lane);
#pragma unroll
      for (int j = 0; j < 4; ++j) {
        bf16x16 b = loadB(Bbuf, KCH, j << 4, kt << 5, lane);
        acc[j] = __builtin_amdgcn_wmma_f32_16x16x32_bf16(false, a, false, b,
                                                         (short)0, acc[j], false, false);
      }
    }
    __syncthreads();
  }

#pragma unroll
  for (int j = 0; j < 4; ++j) {
    const int n = n0base + (j << 4) + (lane & 15);
    const float bias = bc1[n];
#pragma unroll
    for (int r = 0; r < 8; ++r) {
      const int m = m0 + r + (lh << 3);
      C1[(long)m * HC_ + n] = (__bf16)gelu_f(acc[j][r] + bias);
    }
  }
}

// -------- GEMM-G: zg = z@Wg1[H:] + bg1 -> f32 [32768,256] --------
// B staged via Tensor Data Mover (TDM), wave 0 drives the DMA; double-buffered.
__global__ __launch_bounds__(256) void gemm_gate_z(const float* __restrict__ z,
                                                   const __bf16* __restrict__ Wg1t, // [256,2048]
                                                   const float* __restrict__ bg1,
                                                   float* __restrict__ zg) {
  __shared__ __bf16 Bs[2][64 * KCH];        // 2 * 16KB
  const int tid  = threadIdx.x;
  const int lane = tid & 31;
  const int wave = tid >> 5;
  const int mb = blockIdx.x >> 2;           // 0..255
  const int ng = blockIdx.x & 3;            // 0..3
  const int n0base = ng << 6;
  const int m0 = ((mb << 3) + wave) << 4;
  const int row = m0 + (lane & 15);
  const int lh = lane >> 4;
  const int NCHUNK = 8;                     // 1024 / 128

  f32x8 acc[4];
#pragma unroll
  for (int j = 0; j < 4; ++j) acc[j] = zero8();

  auto stage = [&](int c, int buf) {
    const int kc = c << 7;
    // 64 rows x 128 cols bf16 tile from Wg1t[n0base..+63][H_+kc ..+127]
    tdm_load_tile_2d_bf16(&Bs[buf][0],
                          Wg1t + (long)n0base * (2 * H_) + H_ + kc,
                          /*tensor_d0=*/2 * H_, /*tensor_d1=*/HC_,
                          /*tile_d0=*/KCH, /*tile_d1=*/64,
                          /*row_stride=*/2 * H_);
  };

  if (wave == 0) stage(0, 0);
  for (int c = 0; c < NCHUNK; ++c) {
    if (wave == 0) __builtin_amdgcn_s_wait_tensorcnt(0);
    __syncthreads();
    if (c + 1 < NCHUNK && wave == 0) stage(c + 1, (c + 1) & 1);
    const int kc = c << 7;
    const __bf16* Bbuf = &Bs[c & 1][0];
#pragma unroll
    for (int kt = 0; kt < 4; ++kt) {
      bf16x16 a = loadA_f32(z, H_, row, kc + (kt << 5), lane);
#pragma unroll
      for (int j = 0; j < 4; ++j) {
        bf16x16 b = loadB(Bbuf, KCH, j << 4, kt << 5, lane);
        acc[j] = __builtin_amdgcn_wmma_f32_16x16x32_bf16(false, a, false, b,
                                                         (short)0, acc[j], false, false);
      }
    }
    __syncthreads();
  }
#pragma unroll
  for (int j = 0; j < 4; ++j) {
    const int n = n0base + (j << 4) + (lane & 15);
    const float bias = bg1[n];
#pragma unroll
    for (int r = 0; r < 8; ++r) {
      const int m = m0 + r + (lh << 3);
      zg[(long)m * HC_ + n] = acc[j][r] + bias;
    }
  }
}

// -------- GEMM2: cand = u + C1@Wc2 + bc2 -> f32 [32768,1024] --------
// 8 waves share one 64-col N-group; B staged via async global->LDS (K=256 -> 2 chunks).
__global__ __launch_bounds__(256) void gemm_corr2(const __bf16* __restrict__ C1,
                                                  const __bf16* __restrict__ Wc2t, // [1024,256]
                                                  const float* __restrict__ bc2,
                                                  const float* __restrict__ u,
                                                  float* __restrict__ cand) {
  __shared__ __bf16 Bs[2][64 * KCH];        // 2 * 16KB
  const int tid  = threadIdx.x;
  const int lane = tid & 31;
  const int wave = tid >> 5;
  const int mb = blockIdx.x >> 4;           // 0..255
  const int ng = blockIdx.x & 15;           // 0..15
  const int n0base = ng << 6;
  const int m0 = ((mb << 3) + wave) << 4;
  const int row = m0 + (lane & 15);
  const int lh = lane >> 4;
  const int NCHUNK = 2;                     // 256 / 128

  f32x8 acc[4];
#pragma unroll
  for (int j = 0; j < 4; ++j) acc[j] = zero8();

  auto stage = [&](int c, int buf) {
    const int kc = c << 7;
#pragma unroll
    for (int i = 0; i < 4; ++i) {
      const int s  = (i << 8) + tid;
      const int rn = s >> 4;
      const int sk = (s & 15) << 3;
      async_copy_b128(&Bs[buf][rn * KCH + sk],
                      Wc2t + (long)(n0base + rn) * HC_ + kc + sk);
    }
  };

  stage(0, 0);
  for (int c = 0; c < NCHUNK; ++c) {
    wait_async0();
    __syncthreads();
    if (c + 1 < NCHUNK) stage(c + 1, (c + 1) & 1);
    const int kc = c << 7;
    const __bf16* Bbuf = &Bs[c & 1][0];
#pragma unroll
    for (int kt = 0; kt < 4; ++kt) {
      bf16x16 a = loadA_bf16(C1, HC_, row, kc + (kt << 5), lane);
#pragma unroll
      for (int j = 0; j < 4; ++j) {
        bf16x16 b = loadB(Bbuf, KCH, j << 4, kt << 5, lane);
        acc[j] = __builtin_amdgcn_wmma_f32_16x16x32_bf16(false, a, false, b,
                                                         (short)0, acc[j], false, false);
      }
    }
    __syncthreads();
  }
#pragma unroll
  for (int j = 0; j < 4; ++j) {
    const int n = n0base + (j << 4) + (lane & 15);
    const float bias = bc2[n];
#pragma unroll
    for (int r = 0; r < 8; ++r) {
      const int m = m0 + r + (lh << 3);
      cand[(long)m * H_ + n] = u[(long)m * H_ + n] + acc[j][r] + bias;
    }
  }
}

// -------- Sequential gate scan: 1 workgroup, 32 waves, prev in LDS (bf16) --------
// 2 independent WMMA accumulator chains; 2 barriers/step via gsum ping-pong.
__global__ __launch_bounds__(1024) void gate_scan(const float* __restrict__ cand,
                                                  const float* __restrict__ zg,
                                                  const __bf16* __restrict__ Wg1t, // [256,2048], cols 0..1023 = prev half
                                                  const float* __restrict__ Wg2,   // [256]
                                                  const float* __restrict__ bg2,   // [1]
                                                  const unsigned char* __restrict__ vmask,
                                                  float* __restrict__ out) {
  extern __shared__ char smem_raw[];
  __bf16* prev = (__bf16*)smem_raw;                       // 32*1024 bf16 = 64KB
  float* gsum  = (float*)(smem_raw + B_ * H_ * 2);        // 2 x 32 (ping-pong)

  const int tid  = threadIdx.x;
  const int lane = tid & 31;
  const int wave = tid >> 5;
  const int mt = wave & 1;        // 2 M tiles (B=32)
  const int nt = wave >> 1;       // 16 N tiles (HC=256)
  const int m0 = mt << 4;
  const int n0 = nt << 4;
  const int lh = lane >> 4;
  const int ncol = n0 + (lane & 15);
  const float w2 = Wg2[ncol];
  const float bgate = bg2[0];
  const int arow = m0 + (lane & 15);
  const int kh = lh << 3;

#pragma unroll
  for (int i = 0; i < 32; ++i) prev[tid * 32 + i] = (__bf16)0.f;
  if (tid < 2 * B_) gsum[tid] = 0.f;
  __syncthreads();

  for (int t = 0; t < L_; ++t) {
    const int par = t & 1;
    float* gs = gsum + par * B_;
    // acc0 starts at precomputed z-half of the gate pre-activation (bg1 included)
    f32x8 acc0, acc1 = zero8();
#pragma unroll
    for (int r = 0; r < 8; ++r) {
      const int b = m0 + r + (lh << 3);
      acc0[r] = zg[((long)b * L_ + t) * HC_ + ncol];
    }
    // prev @ Wg1[:H], K=1024 -> 32 WMMAs over 2 independent chains
#pragma unroll 2
    for (int kt = 0; kt < 32; kt += 2) {
      const int k0 = kt << 5;
      const __bf16* pa = prev + arow * H_ + k0 + kh;
      bf16x16 a0 = cat8(*(const bf16x8*)(pa), *(const bf16x8*)(pa + 16));
      bf16x16 a1 = cat8(*(const bf16x8*)(pa + 32), *(const bf16x8*)(pa + 48));
      bf16x16 b0 = loadB(Wg1t, 2 * H_, n0, k0, lane);
      bf16x16 b1 = loadB(Wg1t, 2 * H_, n0, k0 + 32, lane);
      acc0 = __builtin_amdgcn_wmma_f32_16x16x32_bf16(false, a0, false, b0,
                                                     (short)0, acc0, false, false);
      acc1 = __builtin_amdgcn_wmma_f32_16x16x32_bf16(false, a1, false, b1,
                                                     (short)0, acc1, false, false);
    }
    f32x8 acc = acc0 + acc1;
    // gelu, weight by Wg2, reduce across the 16 lanes of each row-half
#pragma unroll
    for (int r = 0; r < 8; ++r) {
      float v = gelu_f(acc[r]) * w2;
      v += __shfl_xor(v, 1, 32);
      v += __shfl_xor(v, 2, 32);
      v += __shfl_xor(v, 4, 32);
      v += __shfl_xor(v, 8, 32);
      if ((lane & 15) == 0) atomicAdd(&gs[m0 + r + (lh << 3)], v);
    }
    __syncthreads();
    // blend (sigmoid computed redundantly per thread from raw gsum)
    {
      const int b = tid >> 5;
      const int kbase = (tid & 31) << 5;
      const float g = sigmoid_f(gs[b] + bgate);
      const float mv = vmask[(long)b * L_ + t] ? 1.f : 0.f;
      const float* cp = cand + ((long)b * L_ + t) * H_ + kbase;
      float* op = out + ((long)b * L_ + t) * H_ + kbase;
      __bf16* pp = prev + b * H_ + kbase;
      if (tid < B_) gsum[(1 - par) * B_ + tid] = 0.f;   // clear the other buffer
      if (t + 1 < L_) {
        __builtin_prefetch(cand + ((long)b * L_ + t + 1) * H_ + kbase, 0, 1);
        __builtin_prefetch(zg + ((long)b * L_ + t + 1) * HC_, 0, 1);
      }
#pragma unroll
      for (int i = 0; i < 32; i += 4) {
        f32x4 c = *(const f32x4*)(cp + i);
        f32x4 h;
#pragma unroll
        for (int q = 0; q < 4; ++q) {
          float pv = (float)pp[i + q];
          float hv = (1.f - g) * pv + g * c[q];
          hv = mv * hv + (1.f - mv) * pv;
          h[q] = hv;
        }
        *(f32x4*)(op + i) = h;
#pragma unroll
        for (int q = 0; q < 4; ++q) pp[i + q] = (__bf16)h[q];
      }
    }
    __syncthreads();
  }
}

extern "C" void kernel_launch(void* const* d_in, const int* in_sizes, int n_in,
                              void* d_out, int out_size, void* d_ws, size_t ws_size,
                              hipStream_t stream) {
  const float* u   = (const float*)d_in[0];
  const float* z   = (const float*)d_in[1];
  const unsigned char* vm = (const unsigned char*)d_in[2];
  const float* Wc1 = (const float*)d_in[3];
  const float* bc1 = (const float*)d_in[4];
  const float* Wc2 = (const float*)d_in[5];
  const float* bc2 = (const float*)d_in[6];
  const float* Wg1 = (const float*)d_in[7];
  const float* bg1 = (const float*)d_in[8];
  const float* Wg2 = (const float*)d_in[9];
  const float* bg2 = (const float*)d_in[10];
  float* out = (float*)d_out;

  char* ws = (char*)d_ws;
  size_t off = 0;
  auto take = [&](size_t bytes) -> char* {
    char* p = ws + off;
    off = (off + bytes + 255) & ~(size_t)255;
    return p;
  };
  __bf16* Wc1t = (__bf16*)take((size_t)2 * H_ * HC_ * 2);   // [256,2048] bf16
  __bf16* Wc2t = (__bf16*)take((size_t)H_ * HC_ * 2);       // [1024,256] bf16
  __bf16* Wg1t = (__bf16*)take((size_t)2 * H_ * HC_ * 2);   // [256,2048] bf16
  __bf16* C1   = (__bf16*)take((size_t)B_ * L_ * HC_ * 2);  // [32768,256] bf16
  float*  zg   = (float*) take((size_t)B_ * L_ * HC_ * 4);  // [32768,256] f32
  float*  cand = (float*) take((size_t)B_ * L_ * H_ * 4);   // [32768,1024] f32

  // weight transpose+convert
  f32_to_bf16_t<<<(2 * H_ * HC_ + 255) / 256, 256, 0, stream>>>(Wc1, Wc1t, 2 * H_, HC_);
  f32_to_bf16_t<<<(HC_ * H_ + 255) / 256, 256, 0, stream>>>(Wc2, Wc2t, HC_, H_);
  f32_to_bf16_t<<<(2 * H_ * HC_ + 255) / 256, 256, 0, stream>>>(Wg1, Wg1t, 2 * H_, HC_);

  // parallel GEMMs (8 waves / block, block-shared B in LDS)
  gemm_corr1<<<1024, 256, 0, stream>>>(u, z, Wc1t, bc1, C1);
  gemm_gate_z<<<1024, 256, 0, stream>>>(z, Wg1t, bg1, zg);
  gemm_corr2<<<4096, 256, 0, stream>>>(C1, Wc2t, bc2, u, cand);

  // sequential scan: single WGP, 32 waves, 64KB+256B dynamic LDS
  gate_scan<<<1, 1024, B_ * H_ * 2 + 2 * B_ * 4, stream>>>(cand, zg, Wg1t, Wg2, bg2, vm, out);
}
